// MultiHeadAttention_18098992185396
// MI455X (gfx1250) — matrix operand order repaired and checked by reading, not compiled, not run on hardware
//
#include <hip/hip_runtime.h>
#include <stdint.h>

#define HEADS 6
#define HS 64
#define BATCH 8
#define SEQ 2048
#define CH 384
#define C3 1152
#define MT (BATCH * SEQ)  // 16384 rows

typedef __bf16 bf16;
typedef __attribute__((ext_vector_type(16))) __bf16 v16bf;
typedef __attribute__((ext_vector_type(8))) float v8f;
typedef __attribute__((ext_vector_type(8))) unsigned int v8u;

// fp32 -> bf16 round-to-nearest-even via integer trick (no reliance on native cvt)
static __device__ __forceinline__ bf16 f2bf(float f) {
  unsigned u = __builtin_bit_cast(unsigned, f);
  u = (u + 0x7FFFu + ((u >> 16) & 1u)) >> 16;
  unsigned short s = (unsigned short)u;
  return __builtin_bit_cast(bf16, s);
}

// A-fragment 16x32 k order used here: lane m=lane%16; VGPR v holds
// K = 2v + 16*half .. +1 (contiguous, matches B fragments)
static __device__ __forceinline__ int kmapA(int v, int half) {
  return 2 * v + 16 * half;
}

__global__ void cvt_f32_bf16(const float* __restrict__ s, bf16* __restrict__ d, int n) {
  for (int i = blockIdx.x * blockDim.x + threadIdx.x; i < n; i += gridDim.x * blockDim.x)
    d[i] = f2bf(s[i]);
}

// ---------------------------------------------------------------------------
// Tiled bf16 GEMM: C[M,N] = A[M,K] * B[K,N]; block tile 128x64, BK=32.
// 8 waves (wave32): wave (wm,wn) in 4x2 grid owns a 32x32 accumulator.
// A staged row-major (stride 40 bf16 -> 16B aligned, bank-conflict-free A-frag
// reads), B staged TRANSPOSED (sBt[n][k], stride 34 = 17 dwords, coprime with
// 64 banks) so B-fragment dwords are contiguous K-pairs per the ISA layout.
// ---------------------------------------------------------------------------
template <bool OUT_F32>
__global__ __launch_bounds__(256) void gemm_bf16_nt(
    const bf16* __restrict__ A, const bf16* __restrict__ Bm,
    const float* __restrict__ bias, void* __restrict__ outp,
    int M, int N, int K) {
  __shared__ bf16 sA[128 * 40];
  __shared__ bf16 sBt[64 * 34];
  const int t = threadIdx.x;
  const int lane = t & 31;
  const int w = t >> 5;
  const int wm = w & 3;
  const int wn = w >> 2;
  const int m0 = blockIdx.y * 128;
  const int n0 = blockIdx.x * 64;
  const int nn = lane & 15;  // col for B/C frags, row for A frags
  const int hf = lane >> 4;

  v8f acc[2][2];
#pragma unroll
  for (int i = 0; i < 2; ++i)
#pragma unroll
    for (int jj = 0; jj < 2; ++jj)
#pragma unroll
      for (int r = 0; r < 8; ++r) acc[i][jj][r] = 0.0f;

  for (int kk = 0; kk < K; kk += 32) {
    // stage A 128x32: 512 x 16B chunks, 2 per thread
#pragma unroll
    for (int i = 0; i < 2; ++i) {
      int c = t + i * 256;
      int r = c >> 2, cq = c & 3;
      *(uint4*)(sA + r * 40 + cq * 8) =
          *(const uint4*)(A + (size_t)(m0 + r) * K + kk + cq * 8);
    }
    // stage B 32x64 transposed into sBt[64][34]
#pragma unroll
    for (int i = 0; i < 2; ++i) {
      int c = t + i * 256;
      int r = c >> 4, cq = c & 15;
      uint2 u = *(const uint2*)(Bm + (size_t)(kk + r) * N + n0 + cq * 4);
      const bf16* pe = (const bf16*)&u;
#pragma unroll
      for (int e = 0; e < 4; ++e) sBt[(cq * 4 + e) * 34 + r] = pe[e];
    }
    __syncthreads();

    v8u ua[2], ub[2];
#pragma unroll
    for (int i = 0; i < 2; ++i)
#pragma unroll
      for (int v = 0; v < 8; ++v)
        ua[i][v] = *(const unsigned int*)(sA + (wm * 32 + i * 16 + nn) * 40 + kmapA(v, hf));
#pragma unroll
    for (int jj = 0; jj < 2; ++jj)
#pragma unroll
      for (int v = 0; v < 8; ++v)
        ub[jj][v] = *(const unsigned int*)(sBt + (wn * 32 + jj * 16 + nn) * 34 + 2 * v + 16 * hf);
#pragma unroll
    for (int i = 0; i < 2; ++i)
#pragma unroll
      for (int jj = 0; jj < 2; ++jj)
        acc[i][jj] = __builtin_amdgcn_wmma_f32_16x16x32_bf16(
            false, __builtin_bit_cast(v16bf, ua[i]), false,
            __builtin_bit_cast(v16bf, ub[jj]), (short)0, acc[i][jj], false, false);
    __syncthreads();
  }

  // epilogue: C/D layout -> lane holds rows r+8*half, col nn
#pragma unroll
  for (int i = 0; i < 2; ++i)
#pragma unroll
    for (int jj = 0; jj < 2; ++jj)
#pragma unroll
      for (int r = 0; r < 8; ++r) {
        const int gm = m0 + wm * 32 + i * 16 + r + 8 * hf;
        const int gn = n0 + wn * 32 + jj * 16 + nn;
        const float val = acc[i][jj][r];
        if (OUT_F32)
          ((float*)outp)[(size_t)gm * N + gn] = val + bias[gn];
        else
          ((bf16*)outp)[(size_t)gm * N + gn] = f2bf(val);
      }
}

// ---------------------------------------------------------------------------
// Flash attention: block = 128 queries of one (b,h). 8 waves x 16 query rows.
// K/V tiles (32 keys) staged per j-step; online softmax in exp2 domain.
// qkv layout: [B*T, 3C] bf16 (Q at col h*64, K at C+h*64, V at 2C+h*64).
// ---------------------------------------------------------------------------
__global__ __launch_bounds__(256) void flash_attn(const bf16* __restrict__ qkv,
                                                  bf16* __restrict__ attn) {
  __shared__ bf16 sK[32 * 68];       // [key][d], stride 68 (34 dwords, CF)
  __shared__ bf16 sVt[64 * 34];      // [d][key] transposed, stride 34
  __shared__ bf16 sP[8 * 16 * 34];   // per-wave P scratch (16x32, stride 34)

  const int t = threadIdx.x;
  const int lane = t & 31;
  const int w = t >> 5;
  const int b = blockIdx.x / HEADS;
  const int h = blockIdx.x % HEADS;
  const int q0 = blockIdx.y * 128;
  const int qr0 = q0 + w * 16;       // this wave's first query row
  const int nn = lane & 15;
  const int hf = lane >> 4;

  const float SCALE = 0.125f * 1.44269504088896340736f;  // D^-0.5 * log2(e)
  const float NEG_INF = -__builtin_inff();

  // Q fragments (A-layout), kept in registers for the whole kernel
  v8u aq[2];
  {
    const bf16* qrow = qkv + (size_t)(b * SEQ + qr0 + nn) * C3 + h * HS;
#pragma unroll
    for (int f = 0; f < 2; ++f)
#pragma unroll
      for (int v = 0; v < 8; ++v)
        aq[f][v] = *(const unsigned int*)(qrow + f * 32 + kmapA(v, hf));
  }

  v8f o[4];
#pragma unroll
  for (int f = 0; f < 4; ++f)
#pragma unroll
    for (int r = 0; r < 8; ++r) o[f][r] = 0.0f;
  float mst[8], lst[8];
#pragma unroll
  for (int r = 0; r < 8; ++r) { mst[r] = NEG_INF; lst[r] = 0.0f; }

  const int jmax = q0 / 32 + 4;  // keys needed: up to q0+127
  bf16* pscr = sP + w * (16 * 34);

  for (int j = 0; j < jmax; ++j) {
    const int k0 = j * 32;
    // stage K tile [32 x 64]
#pragma unroll
    for (int i = 0; i < 2; ++i) {
      int c = t + i * 256;
      int r = c >> 4, cq = c & 15;
      *(uint2*)(sK + r * 68 + cq * 4) =
          *(const uint2*)(qkv + (size_t)(b * SEQ + k0 + r) * C3 + CH + h * HS + cq * 4);
    }
    // stage V tile transposed -> sVt[d][key]
#pragma unroll
    for (int i = 0; i < 2; ++i) {
      int c = t + i * 256;
      int r = c >> 4, cq = c & 15;
      uint2 u = *(const uint2*)(qkv + (size_t)(b * SEQ + k0 + r) * C3 + 2 * CH + h * HS + cq * 4);
      const bf16* pe = (const bf16*)&u;
#pragma unroll
      for (int e = 0; e < 4; ++e) sVt[(cq * 4 + e) * 34 + r] = pe[e];
    }
    __syncthreads();

    if (k0 <= qr0 + 15) {  // wave-uniform: skip fully-masked future tiles
      // S(16x32) = Q(16x64) * K^T : two 16x16 frags, 2 wmma each over D
      v8f s[2];
#pragma unroll
      for (int kf = 0; kf < 2; ++kf) {
        v8f cacc;
#pragma unroll
        for (int r = 0; r < 8; ++r) cacc[r] = 0.0f;
#pragma unroll
        for (int c = 0; c < 2; ++c) {
          v8u bk;  // B elem (d,n) = K[kf*16+n][c*32+d] -> contiguous pairs
#pragma unroll
          for (int v = 0; v < 8; ++v)
            bk[v] = *(const unsigned int*)(sK + (kf * 16 + nn) * 68 + c * 32 + 2 * v + 16 * hf);
          cacc = __builtin_amdgcn_wmma_f32_16x16x32_bf16(
              false, __builtin_bit_cast(v16bf, aq[c]), false,
              __builtin_bit_cast(v16bf, bk), (short)0, cacc, false, false);
        }
        s[kf] = cacc;
      }
      // online softmax (lane holds rows r+8*hf, one col nn per half)
      float alpha[8];
#pragma unroll
      for (int r = 0; r < 8; ++r) {
        const int qidx = qr0 + r + 8 * hf;
        float v0 = s[0][r] * SCALE;
        float v1 = s[1][r] * SCALE;
        if (k0 + nn > qidx) v0 = NEG_INF;        // causal mask
        if (k0 + 16 + nn > qidx) v1 = NEG_INF;
        float mx = fmaxf(v0, v1);
#pragma unroll
        for (int d = 1; d < 16; d <<= 1) mx = fmaxf(mx, __shfl_xor(mx, d, 32));
        const float mnew = fmaxf(mst[r], mx);
        const float a = exp2f(mst[r] - mnew);
        const float p0 = exp2f(v0 - mnew);
        const float p1 = exp2f(v1 - mnew);
        s[0][r] = p0;
        s[1][r] = p1;
        float sum = p0 + p1;
#pragma unroll
        for (int d = 1; d < 16; d <<= 1) sum += __shfl_xor(sum, d, 32);
        lst[r] = lst[r] * a + sum;
        mst[r] = mnew;
        alpha[r] = a;
      }
#pragma unroll
      for (int f = 0; f < 4; ++f)
#pragma unroll
        for (int r = 0; r < 8; ++r) o[f][r] *= alpha[r];

      // P (f32 C-layout) -> bf16 A-layout via per-wave LDS scratch
#pragma unroll
      for (int kf = 0; kf < 2; ++kf)
#pragma unroll
        for (int r = 0; r < 8; ++r)
          pscr[(r + 8 * hf) * 34 + kf * 16 + nn] = f2bf(s[kf][r]);
      asm volatile("s_wait_dscnt 0" ::: "memory");  // LDS in-order per wave
      v8u ap;
#pragma unroll
      for (int v = 0; v < 8; ++v)
        ap[v] = *(const unsigned int*)(pscr + nn * 34 + kmapA(v, hf));

      // O(16x64) += P(16x32) * V(32x64): 4 wmma over d-slices
#pragma unroll
      for (int f = 0; f < 4; ++f) {
        v8u bv;  // B elem (k,n) = V[k][f*16+n] = sVt[f*16+n][k], contiguous
#pragma unroll
        for (int v = 0; v < 8; ++v)
          bv[v] = *(const unsigned int*)(sVt + (f * 16 + nn) * 34 + 2 * v + 16 * hf);
        o[f] = __builtin_amdgcn_wmma_f32_16x16x32_bf16(
            false, __builtin_bit_cast(v16bf, ap), false,
            __builtin_bit_cast(v16bf, bv), (short)0, o[f], false, false);
      }
    }
    __syncthreads();
  }

  // epilogue: O /= l, store bf16 attn[B*T, C] at col h*64
#pragma unroll
  for (int r = 0; r < 8; ++r) {
    const float inv = 1.0f / lst[r];
    bf16* orow = attn + (size_t)(b * SEQ + qr0 + r + 8 * hf) * CH + h * HS;
#pragma unroll
    for (int f = 0; f < 4; ++f) orow[f * 16 + nn] = f2bf(o[f][r] * inv);
  }
}

extern "C" void kernel_launch(void* const* d_in, const int* in_sizes, int n_in,
                              void* d_out, int out_size, void* d_ws, size_t ws_size,
                              hipStream_t stream) {
  (void)in_sizes; (void)n_in; (void)out_size; (void)ws_size;
  const float* x = (const float*)d_in[0];
  const float* w_qkv = (const float*)d_in[1];
  const float* w_proj = (const float*)d_in[2];
  const float* b_proj = (const float*)d_in[3];
  float* out = (float*)d_out;

  char* ws = (char*)d_ws;
  size_t off = 0;
  auto carve = [&](size_t bytes) -> char* {
    char* p = ws + off;
    off += (bytes + 255) & ~(size_t)255;
    return p;
  };
  bf16* xb     = (bf16*)carve((size_t)MT * CH * 2);   // 12.6 MB
  bf16* wqkvb  = (bf16*)carve((size_t)CH * C3 * 2);   // 0.9 MB
  bf16* wprojb = (bf16*)carve((size_t)CH * CH * 2);   // 0.3 MB
  bf16* qkvb   = (bf16*)carve((size_t)MT * C3 * 2);   // 37.7 MB (fits in L2)
  bf16* attnb  = (bf16*)carve((size_t)MT * CH * 2);   // 12.6 MB

  cvt_f32_bf16<<<512, 256, 0, stream>>>(x, xb, MT * CH);
  cvt_f32_bf16<<<256, 256, 0, stream>>>(w_qkv, wqkvb, CH * C3);
  cvt_f32_bf16<<<64, 256, 0, stream>>>(w_proj, wprojb, CH * CH);

  // qkv = x @ w_qkv  (bf16 out)
  gemm_bf16_nt<false><<<dim3(C3 / 64, MT / 128), 256, 0, stream>>>(
      xb, wqkvb, nullptr, (void*)qkvb, MT, C3, CH);

  // fused causal attention per (b,h)
  flash_attn<<<dim3(BATCH * HEADS, SEQ / 128), 256, 0, stream>>>(qkvb, attnb);

  // out = attn @ w_proj + b_proj (f32 out)
  gemm_bf16_nt<true><<<dim3(CH / 64, MT / 128), 256, 0, stream>>>(
      attnb, wprojb, b_proj, (void*)out, MT, CH, CH);
}
